// PureMambaClassifier_28913719837240
// MI455X (gfx1250) — matrix operand order repaired and checked by reading, not compiled, not run on hardware
//
#include <hip/hip_runtime.h>
#include <hip/hip_bf16.h>
#include <math.h>

// ---------------------------------------------------------------- constants
#define D_MODEL  512
#define D_STATE  16
#define D_CONV   4
#define D_INNER  1024
#define DT_RANK  32
#define N_LAYERS 4
#define BB       8
#define SS       2048
#define ROWS     (BB * SS)          // 16384
#define XDBL_W   (DT_RANK + 2 * D_STATE)   // 64

typedef __attribute__((ext_vector_type(16))) __bf16 bf16x16;
typedef __attribute__((ext_vector_type(8)))  float  f32x8;

// ---------------------------------------------------------------- CDNA5 async copy helpers
// GLOBAL_LOAD_ASYNC_TO_LDS_B128: memory -> LDS, no VGPR data, tracked by ASYNCcnt.
__device__ __forceinline__ void async_copy_b128(unsigned lds_off, const void* gptr) {
    asm volatile("global_load_async_to_lds_b128 %0, %1, off"
                 :: "v"(lds_off), "v"(gptr) : "memory");
}
__device__ __forceinline__ void wait_async0() {
    asm volatile("s_wait_asynccnt 0" ::: "memory");
}

// ---------------------------------------------------------------- GEMM (C = A * W^T)
// A: (M,K) bf16, leading dim lda.  W: (N,K) bf16, leading dim ldw.  C: (M,N) f32.
// mode 0: plain store; mode 1: softplus(C + bias[n]).  Cbf: optional bf16 shadow.
// Block = 256 threads = 8 waves arranged 4(m) x 2(n).
// Block tile: 128 x (32*NFRAG).  Wave tile: 32 x (16*NFRAG).
// Double-buffered LDS staging via global_load_async_to_lds_b128 + s_wait_asynccnt.
template <int NFRAG>
__global__ __launch_bounds__(256)
void gemm_bf16_wmma(const __bf16* __restrict__ A, int lda,
                    const __bf16* __restrict__ W, int ldw,
                    float* __restrict__ C, int ldc,
                    __bf16* __restrict__ Cbf,
                    const float* __restrict__ bias,
                    int K, int mode) {
    constexpr int BM = 128;
    constexpr int BN = 2 * 16 * NFRAG;          // 128 or 64
    __shared__ __bf16 sA[2][BM * 32];
    __shared__ __bf16 sB[2][BN * 32];

    const int t    = threadIdx.x;
    const int lane = t & 31;
    const int wave = t >> 5;
    const int wm   = wave >> 1;                 // 0..3
    const int wn   = wave & 1;                  // 0..1
    const int rowBlk = blockIdx.y * BM;
    const int colBlk = blockIdx.x * BN;

    // LDS byte offsets for the asm path (generic->LDS offset = low 32 bits)
    const unsigned ldsA[2] = { (unsigned)(uintptr_t)&sA[0][0], (unsigned)(uintptr_t)&sA[1][0] };
    const unsigned ldsB[2] = { (unsigned)(uintptr_t)&sB[0][0], (unsigned)(uintptr_t)&sB[1][0] };

    // per-thread staging assignment: row r (64B of K-chunk), half hb
    const int r  = t >> 1;                      // 0..127
    const int hb = (t & 1) * 32;                // byte offset within 64B row

    auto stage = [&](int k0, int buf) {
        unsigned la = ldsA[buf] + (unsigned)(r * 64 + hb);
        const char* ga = (const char*)(A + (size_t)(rowBlk + r) * lda + k0) + hb;
        async_copy_b128(la, ga);
        async_copy_b128(la + 16, ga + 16);
        if (r < BN) {
            unsigned lb = ldsB[buf] + (unsigned)(r * 64 + hb);
            const char* gb = (const char*)(W + (size_t)(colBlk + r) * ldw + k0) + hb;
            async_copy_b128(lb, gb);
            async_copy_b128(lb + 16, gb + 16);
        }
    };

    f32x8 acc[2][NFRAG] = {};

    const int nk = K >> 5;                      // K chunks of 32
    stage(0, 0);
    wait_async0();
    __syncthreads();

    for (int kc = 0; kc < nk; ++kc) {
        if (kc + 1 < nk) stage((kc + 1) << 5, (kc + 1) & 1);

        const __bf16* a_base = &sA[kc & 1][0];
        const __bf16* b_base = &sB[kc & 1][0];

        // A fragments (16x32 MxK): lane L<16 -> row M=L, K={0..7,16..23};
        // lane L>=16 -> row M=L-16, K={8..15,24..31}.  (ISA 7.12.2)
        bf16x16 af[2];
#pragma unroll
        for (int mt = 0; mt < 2; ++mt) {
            const __bf16* p = a_base + (wm * 32 + mt * 16 + (lane & 15)) * 32 + ((lane >> 4) << 3);
            bf16x16 f;
#pragma unroll
            for (int i = 0; i < 8; ++i) f[i] = p[i];
#pragma unroll
            for (int i = 0; i < 8; ++i) f[8 + i] = p[16 + i];
            af[mt] = f;
        }
        // B fragments (32x16 KxN): lane L<16 -> col N=L, K={0..7,16..23}; L>=16 -> K={8..15,24..31}.
#pragma unroll
        for (int nf = 0; nf < NFRAG; ++nf) {
            const __bf16* p = b_base + (wn * 16 * NFRAG + nf * 16 + (lane & 15)) * 32 + ((lane >> 4) << 3);
            bf16x16 bfr;
#pragma unroll
            for (int i = 0; i < 8; ++i) { bfr[i] = p[i]; bfr[8 + i] = p[16 + i]; }
#pragma unroll
            for (int mt = 0; mt < 2; ++mt)
                acc[mt][nf] = __builtin_amdgcn_wmma_f32_16x16x32_bf16(
                    false, af[mt], false, bfr, (short)0, acc[mt][nf], false, false);
        }
        wait_async0();      // own asyncs for next stage complete
        __syncthreads();    // everyone's staged data visible / buffer reuse safe
    }

    // D layout: element r of lane L -> M = r + 8*(L>>4), N = L&15
    const int hi = (lane >> 4) * 8;
    const int nn = lane & 15;
#pragma unroll
    for (int mt = 0; mt < 2; ++mt) {
#pragma unroll
        for (int nf = 0; nf < NFRAG; ++nf) {
            const int col = colBlk + wn * 16 * NFRAG + nf * 16 + nn;
#pragma unroll
            for (int rr = 0; rr < 8; ++rr) {
                const int row = rowBlk + wm * 32 + mt * 16 + hi + rr;
                float v = acc[mt][nf][rr];
                if (mode == 1) {                      // softplus(v + bias)
                    v += bias[col];
                    v = (v > 20.f) ? v : log1pf(__expf(v));
                }
                const size_t idx = (size_t)row * ldc + col;
                C[idx] = v;
                if (Cbf) Cbf[idx] = (__bf16)v;
            }
        }
    }
}

// ---------------------------------------------------------------- misc kernels
__global__ void convert_f32_to_bf16(const float* __restrict__ in,
                                    __bf16* __restrict__ out, size_t n) {
    size_t i = (size_t)blockIdx.x * blockDim.x + threadIdx.x;
    for (; i < n; i += (size_t)gridDim.x * blockDim.x) out[i] = (__bf16)in[i];
}

__global__ void embed_kernel(const int* __restrict__ ids,
                             const float* __restrict__ emb,
                             const float* __restrict__ pos,
                             float* __restrict__ x, __bf16* __restrict__ xb) {
    size_t i = (size_t)blockIdx.x * blockDim.x + threadIdx.x;
    if (i >= (size_t)ROWS * D_MODEL) return;
    int d = (int)(i % D_MODEL);
    size_t bs = i / D_MODEL;
    int s = (int)(bs % SS);
    int tok = ids[bs];
    float v = emb[(size_t)tok * D_MODEL + d] + pos[(size_t)s * D_MODEL + d];
    x[i] = v;
    xb[i] = (__bf16)v;
}

// depthwise causal conv (width 4) + bias + SiLU over xz[..., :D_INNER]
__global__ void conv_silu_kernel(const float* __restrict__ xz,
                                 const float* __restrict__ cw,
                                 const float* __restrict__ cb,
                                 float* __restrict__ xc, __bf16* __restrict__ xcb) {
    size_t i = (size_t)blockIdx.x * blockDim.x + threadIdx.x;
    if (i >= (size_t)ROWS * D_INNER) return;
    int d = (int)(i % D_INNER);
    size_t bs = i / D_INNER;
    int s = (int)(bs % SS);
    size_t b = bs / SS;
    const float* xi = xz + b * (size_t)SS * (2 * D_INNER);
    float acc = cb[d];
#pragma unroll
    for (int w = 0; w < D_CONV; ++w) {
        int sp = s - (D_CONV - 1) + w;
        if (sp >= 0) acc += xi[(size_t)sp * (2 * D_INNER) + d] * cw[d * D_CONV + w];
    }
    float sv = acc / (1.f + __expf(-acc));   // SiLU
    xc[i] = sv;
    xcb[i] = (__bf16)sv;
}

// selective scan: one thread per (b, d); 16 states in registers.
__global__ void scan_kernel(const float* __restrict__ xc,
                            const float* __restrict__ dt,
                            const float* __restrict__ xdbl,
                            const float* __restrict__ A_log,
                            const float* __restrict__ Dp,
                            float* __restrict__ y) {
    int tid = blockIdx.x * blockDim.x + threadIdx.x;
    if (tid >= BB * D_INNER) return;
    int b = tid / D_INNER;
    int d = tid % D_INNER;
    float Arow[D_STATE];
#pragma unroll
    for (int n = 0; n < D_STATE; ++n) Arow[n] = -__expf(A_log[(size_t)d * D_STATE + n]);
    const float Dv = Dp[d];
    float h[D_STATE];
#pragma unroll
    for (int n = 0; n < D_STATE; ++n) h[n] = 0.f;

    const size_t rowb = (size_t)b * SS;
    for (int s = 0; s < SS; ++s) {
        const size_t r = rowb + s;
        const float dtv = dt[r * D_INNER + d];
        const float xv  = xc[r * D_INNER + d];
        const float* bc = xdbl + r * XDBL_W + DT_RANK;   // B then C
        float yv = 0.f;
#pragma unroll
        for (int n = 0; n < D_STATE; ++n) {
            float dA = __expf(dtv * Arow[n]);
            h[n] = dA * h[n] + dtv * bc[n] * xv;
            yv += h[n] * bc[D_STATE + n];
        }
        y[r * D_INNER + d] = yv + Dv * xv;
    }
}

// y *= silu(z);  z = xz[..., D_INNER:]
__global__ void gate_kernel(const float* __restrict__ xz,
                            float* __restrict__ y, __bf16* __restrict__ yb) {
    size_t i = (size_t)blockIdx.x * blockDim.x + threadIdx.x;
    if (i >= (size_t)ROWS * D_INNER) return;
    int d = (int)(i % D_INNER);
    size_t bs = i / D_INNER;
    float z = xz[bs * (2 * D_INNER) + D_INNER + d];
    float v = y[i] * (z / (1.f + __expf(-z)));
    y[i] = v;
    yb[i] = (__bf16)v;
}

// x = LayerNorm(src (+ res)) * g + b ; optionally write bf16 copy
__global__ void layernorm_kernel(const float* __restrict__ src,
                                 const float* __restrict__ res,
                                 const float* __restrict__ g,
                                 const float* __restrict__ bta,
                                 float* __restrict__ xo, __bf16* __restrict__ xbf,
                                 int hasRes, int writeBf) {
    const int row = blockIdx.x;
    const int t = threadIdx.x;   // 256 threads, 2 cols each
    __shared__ float red[256];
    const size_t base = (size_t)row * D_MODEL;
    float v0 = src[base + t];
    float v1 = src[base + 256 + t];
    if (hasRes) { v0 += res[base + t]; v1 += res[base + 256 + t]; }
    red[t] = v0 + v1;
    __syncthreads();
    for (int o = 128; o > 0; o >>= 1) { if (t < o) red[t] += red[t + o]; __syncthreads(); }
    const float mean = red[0] * (1.f / D_MODEL);
    __syncthreads();
    float d0 = v0 - mean, d1 = v1 - mean;
    red[t] = d0 * d0 + d1 * d1;
    __syncthreads();
    for (int o = 128; o > 0; o >>= 1) { if (t < o) red[t] += red[t + o]; __syncthreads(); }
    const float rstd = rsqrtf(red[0] * (1.f / D_MODEL) + 1e-5f);
    float o0 = d0 * rstd * g[t] + bta[t];
    float o1 = d1 * rstd * g[256 + t] + bta[256 + t];
    xo[base + t] = o0;
    xo[base + 256 + t] = o1;
    if (writeBf) { xbf[base + t] = (__bf16)o0; xbf[base + 256 + t] = (__bf16)o1; }
}

// masked mean pool: thread per (b, d)
__global__ void pool_kernel(const float* __restrict__ x,
                            const int* __restrict__ mask,
                            float* __restrict__ pooled) {
    int tid = blockIdx.x * blockDim.x + threadIdx.x;
    if (tid >= BB * D_MODEL) return;
    int b = tid / D_MODEL;
    int d = tid % D_MODEL;
    float num = 0.f, den = 0.f;
    for (int s = 0; s < SS; ++s) {
        float m = (float)mask[(size_t)b * SS + s];
        num += x[((size_t)b * SS + s) * D_MODEL + d] * m;
        den += m;
    }
    pooled[(size_t)b * D_MODEL + d] = num / fmaxf(den, 1e-9f);
}

__global__ void cls1_kernel(const float* __restrict__ pooled,
                            const float* __restrict__ w1,
                            const float* __restrict__ b1,
                            float* __restrict__ h) {
    int tid = blockIdx.x * blockDim.x + threadIdx.x;
    if (tid >= BB * D_MODEL) return;
    int b = tid / D_MODEL;
    int j = tid % D_MODEL;
    float acc = b1[j];
    for (int d = 0; d < D_MODEL; ++d)
        acc += pooled[(size_t)b * D_MODEL + d] * w1[(size_t)j * D_MODEL + d];
    h[tid] = 0.5f * acc * (1.f + erff(acc * 0.70710678118654752f));   // exact GELU
}

__global__ void cls2_kernel(const float* __restrict__ h,
                            const float* __restrict__ w2,
                            const float* __restrict__ b2,
                            float* __restrict__ out, int numLabels) {
    int tid = blockIdx.x * blockDim.x + threadIdx.x;
    if (tid >= BB * numLabels) return;
    int b = tid / numLabels;
    int c = tid % numLabels;
    float acc = b2[c];
    for (int d = 0; d < D_MODEL; ++d)
        acc += h[(size_t)b * D_MODEL + d] * w2[(size_t)c * D_MODEL + d];
    out[tid] = acc;
}

// ---------------------------------------------------------------- host launch
extern "C" void kernel_launch(void* const* d_in, const int* in_sizes, int n_in,
                              void* d_out, int out_size, void* d_ws, size_t ws_size,
                              hipStream_t stream) {
    (void)in_sizes; (void)n_in; (void)out_size; (void)ws_size;

    const int*   ids    = (const int*)d_in[0];
    const int*   mask   = (const int*)d_in[1];
    const float* emb    = (const float*)d_in[2];
    const float* pos    = (const float*)d_in[3];
    const float* in_w   = (const float*)d_in[4];
    const float* conv_w = (const float*)d_in[5];
    const float* conv_b = (const float*)d_in[6];
    const float* x_w    = (const float*)d_in[7];
    const float* dt_w   = (const float*)d_in[8];
    const float* dt_b   = (const float*)d_in[9];
    const float* A_log  = (const float*)d_in[10];
    const float* Dp     = (const float*)d_in[11];
    const float* out_w  = (const float*)d_in[12];
    const float* ln_g   = (const float*)d_in[13];
    const float* ln_b   = (const float*)d_in[14];
    const float* fn_g   = (const float*)d_in[15];
    const float* fn_b   = (const float*)d_in[16];
    const float* cls_w1 = (const float*)d_in[17];
    const float* cls_b1 = (const float*)d_in[18];
    const float* cls_w2 = (const float*)d_in[19];
    const float* cls_b2 = (const float*)d_in[20];

    // workspace carve-out
    char* p = (char*)d_ws;
    auto carve = [&](size_t bytes) { char* q = p; p += (bytes + 255) & ~(size_t)255; return q; };
    float*  X      = (float*)carve((size_t)ROWS * D_MODEL * 4);
    __bf16* XB     = (__bf16*)carve((size_t)ROWS * D_MODEL * 2);
    float*  XZ     = (float*)carve((size_t)ROWS * 2 * D_INNER * 4);
    float*  XC     = (float*)carve((size_t)ROWS * D_INNER * 4);
    __bf16* XCB    = (__bf16*)carve((size_t)ROWS * D_INNER * 2);
    float*  XDBL   = (float*)carve((size_t)ROWS * XDBL_W * 4);
    __bf16* XDBLB  = (__bf16*)carve((size_t)ROWS * XDBL_W * 2);
    float*  DT     = (float*)carve((size_t)ROWS * D_INNER * 4);
    float*  Y      = (float*)carve((size_t)ROWS * D_INNER * 4);
    __bf16* YB     = (__bf16*)carve((size_t)ROWS * D_INNER * 2);
    float*  OUT    = (float*)carve((size_t)ROWS * D_MODEL * 4);
    __bf16* WIN    = (__bf16*)carve((size_t)N_LAYERS * 2 * D_INNER * D_MODEL * 2);
    __bf16* WX     = (__bf16*)carve((size_t)N_LAYERS * XDBL_W * D_INNER * 2);
    __bf16* WDT    = (__bf16*)carve((size_t)N_LAYERS * D_INNER * DT_RANK * 2);
    __bf16* WOUT   = (__bf16*)carve((size_t)N_LAYERS * D_MODEL * D_INNER * 2);
    float*  POOLED = (float*)carve((size_t)BB * D_MODEL * 4);
    float*  H      = (float*)carve((size_t)BB * D_MODEL * 4);

    // weight conversion (f32 -> bf16), once per launch (deterministic)
    convert_f32_to_bf16<<<1024, 256, 0, stream>>>(in_w,  WIN,  (size_t)N_LAYERS * 2 * D_INNER * D_MODEL);
    convert_f32_to_bf16<<<256,  256, 0, stream>>>(x_w,   WX,   (size_t)N_LAYERS * XDBL_W * D_INNER);
    convert_f32_to_bf16<<<256,  256, 0, stream>>>(dt_w,  WDT,  (size_t)N_LAYERS * D_INNER * DT_RANK);
    convert_f32_to_bf16<<<1024, 256, 0, stream>>>(out_w, WOUT, (size_t)N_LAYERS * D_MODEL * D_INNER);

    // embedding + position
    {
        size_t n = (size_t)ROWS * D_MODEL;
        embed_kernel<<<(unsigned)((n + 255) / 256), 256, 0, stream>>>(ids, emb, pos, X, XB);
    }

    const size_t nInner = (size_t)ROWS * D_INNER;
    const unsigned gInner = (unsigned)((nInner + 255) / 256);

    for (int l = 0; l < N_LAYERS; ++l) {
        const __bf16* win  = WIN  + (size_t)l * 2 * D_INNER * D_MODEL;
        const __bf16* wx   = WX   + (size_t)l * XDBL_W * D_INNER;
        const __bf16* wdt  = WDT  + (size_t)l * D_INNER * DT_RANK;
        const __bf16* wout = WOUT + (size_t)l * D_MODEL * D_INNER;
        const float*  cw   = conv_w + (size_t)l * D_INNER * D_CONV;
        const float*  cb   = conv_b + (size_t)l * D_INNER;
        const float*  dtb  = dt_b   + (size_t)l * D_INNER;
        const float*  al   = A_log  + (size_t)l * D_INNER * D_STATE;
        const float*  dp   = Dp     + (size_t)l * D_INNER;
        const float*  lg   = ln_g   + (size_t)l * D_MODEL;
        const float*  lb   = ln_b   + (size_t)l * D_MODEL;

        // in_proj: (16384,512) x (2048,512)^T -> XZ (16384,2048)
        gemm_bf16_wmma<4><<<dim3(2 * D_INNER / 128, ROWS / 128), 256, 0, stream>>>(
            XB, D_MODEL, win, D_MODEL, XZ, 2 * D_INNER, nullptr, nullptr, D_MODEL, 0);

        // depthwise conv + SiLU
        conv_silu_kernel<<<gInner, 256, 0, stream>>>(XZ, cw, cb, XC, XCB);

        // x_proj: (16384,1024) x (64,1024)^T -> XDBL (16384,64) (+bf16 shadow)
        gemm_bf16_wmma<2><<<dim3(XDBL_W / 64, ROWS / 128), 256, 0, stream>>>(
            XCB, D_INNER, wx, D_INNER, XDBL, XDBL_W, XDBLB, nullptr, D_INNER, 0);

        // dt_proj + softplus: (16384, first 32 of 64) x (1024,32)^T -> DT (16384,1024)
        gemm_bf16_wmma<4><<<dim3(D_INNER / 128, ROWS / 128), 256, 0, stream>>>(
            XDBLB, XDBL_W, wdt, DT_RANK, DT, D_INNER, nullptr, dtb, DT_RANK, 1);

        // selective scan
        scan_kernel<<<(BB * D_INNER + 255) / 256, 256, 0, stream>>>(XC, DT, XDBL, al, dp, Y);

        // y *= silu(z)
        gate_kernel<<<gInner, 256, 0, stream>>>(XZ, Y, YB);

        // out_proj: (16384,1024) x (512,1024)^T -> OUT (16384,512)
        gemm_bf16_wmma<4><<<dim3(D_MODEL / 128, ROWS / 128), 256, 0, stream>>>(
            YB, D_INNER, wout, D_INNER, OUT, D_MODEL, nullptr, nullptr, D_INNER, 0);

        // x = LN(OUT + X)
        layernorm_kernel<<<ROWS, 256, 0, stream>>>(OUT, X, lg, lb, X, XB, 1, 1);
    }

    // final LN (no residual, no bf16 needed)
    layernorm_kernel<<<ROWS, 256, 0, stream>>>(X, nullptr, fn_g, fn_b, X, XB, 0, 0);

    // masked mean pool + classifier head
    pool_kernel<<<(BB * D_MODEL + 255) / 256, 256, 0, stream>>>(X, mask, POOLED);
    cls1_kernel<<<(BB * D_MODEL + 255) / 256, 256, 0, stream>>>(POOLED, cls_w1, cls_b1, H);
    cls2_kernel<<<1, 64, 0, stream>>>(H, cls_w2, cls_b2, (float*)d_out, 2);
}
